// RelationLayer_69054484185608
// MI455X (gfx1250) — matrix operand order repaired and checked by reading, not compiled, not run on hardware
//
#include <hip/hip_runtime.h>
#include <hip/hip_bf16.h>

typedef _Float16 v16h __attribute__((ext_vector_type(16)));
typedef _Float16 h2   __attribute__((ext_vector_type(2)));
typedef float    v8f  __attribute__((ext_vector_type(8)));

// f32 leaky-relu for the final store: 0.505*v + 0.495*|v| (no fmax canonicalize)
__device__ __forceinline__ float lrelu(float v) {
    return fmaf(0.505f, v, fabsf(0.495f * v));
}

union HU  { h2 h; int i; };
__device__ __forceinline__ int h2i(h2 v)  { HU u; u.h = v; return u.i; }
__device__ __forceinline__ h2  ih2(int v) { HU u; u.i = v; return u.h; }
union BU  { v16h v; int w[8]; };

// Packed f16 leaky-relu on a cvt_pk'd pair: v_pk_mul_f16 + v_pk_max_num_f16.
__device__ __forceinline__ h2 lrelu_pk(float va, float vb) {
    h2 t; t[0] = (_Float16)va; t[1] = (_Float16)vb;      // v_cvt_pk_f16_f32
    h2 s; s[0] = (_Float16)0.01f; s[1] = (_Float16)0.01f;
    return __builtin_elementwise_max(t, t * s);
}

// One wave = 16 batch elements. All three Linear layers run transposed on
// v_wmma_f32_16x16x32_f16: D[M=out_feat][N=object/batch] = W(A) x data(B) + bias(C).
//
// Split-K trick: B column n gets K=0..7 (e0..7) from lane n and K=8..15 (e0..7) from
// lane n+16. We place the weight columns the *producing* lane half already owns
// at the matching A-K range (zeros elsewhere), so chained GEMMs need no
// transpose, no shuffle, and no zero-masking (0-weight K slots make the B side
// don't-care as long as it is finite).
__global__ __launch_bounds__(256) void relation_wmma_kernel(
    const float* __restrict__ x,
    const float* __restrict__ own_emb,
    const float* __restrict__ agent_emb,
    const float* __restrict__ subgoal_emb,
    const float* __restrict__ goal_emb,
    const float* __restrict__ rel_w0, const float* __restrict__ rel_b0,
    const float* __restrict__ rel_w1, const float* __restrict__ rel_b1,
    const float* __restrict__ mlp_w0, const float* __restrict__ mlp_b0,
    float* __restrict__ out, int ntiles)
{
    const int lane = threadIdx.x & 31;
    const int wave = threadIdx.x >> 5;
    const int tile = blockIdx.x * (blockDim.x >> 5) + wave;
    if (tile >= ntiles) return;

    const int  m   = lane & 15;
    const bool lo  = lane < 16;
    const int  off = lo ? 0 : 8;

    // ---- A operands (weights). Elements 8..15 of every lane all zero. ----
    v16h a1, a2, a3;
    #pragma unroll
    for (int e = 0; e < 16; ++e) {
        a1[e] = (_Float16)0.f; a2[e] = (_Float16)0.f; a3[e] = (_Float16)0.f;
    }
    {
        // GEMM1 contraction: low lanes e0..6 -> combined[0..6] (own half),
        //                    high lanes e0..6 -> combined[7..13] (object half).
        #pragma unroll
        for (int e = 0; e < 7; ++e) {
            a1[e]     = (_Float16)rel_w0[m * 14 + (lo ? 0 : 7) + e];
        }
        // GEMM2/3 contraction: low lanes e0..7 -> feats 0..7, high lanes e0..7 -> feats 8..15.
        #pragma unroll
        for (int e = 0; e < 8; ++e) {
            a2[e]     = (_Float16)rel_w1[m * 16 + off + e];
            a3[e]     = (_Float16)mlp_w0[m * 16 + off + e];
        }
    }

    // ---- C operands: bias[M], M = v + (lo?0:8) ----
    v8f c1, c2, c3;
    #pragma unroll
    for (int v = 0; v < 8; ++v) {
        c1[v] = rel_b0[off + v];
        c2[v] = rel_b1[off + v];
        c3[v] = mlp_b0[off + v];
    }

    // ---- B1: lane n supplies own half (coords+own_emb), lane n+16 supplies
    //      object half (obj coords + emb). e3..6 are batch-invariant. ----
    v16h b1;
    #pragma unroll
    for (int e = 0; e < 16; ++e) b1[e] = (_Float16)0.f;
    const float* ep = lo ? own_emb
                    : (m == 0)  ? own_emb
                    : (m == 1)  ? agent_emb
                    : (m == 15) ? goal_emb : subgoal_emb;
    #pragma unroll
    for (int i = 0; i < 4; ++i) b1[3 + i] = (_Float16)ep[i];

    BU b2, b3;
    #pragma unroll
    for (int i = 0; i < 8; ++i) { b2.w[i] = 0; b3.w[i] = 0; }

    const int  qoff = lo ? 0 : 3 * m;   // which 3 coords this lane loads
    const long base = (long)tile * 16;

    #pragma unroll 4
    for (int j = 0; j < 16; ++j) {
        const float* xp = x + (base + j) * 48 + qoff;
        b1[0] = (_Float16)xp[0];
        b1[1] = (_Float16)xp[1];
        b1[2] = (_Float16)xp[2];

        // GEMM1: h1^T[feat][obj]
        v8f d1 = __builtin_amdgcn_wmma_f32_16x16x32_f16(
            false, a1, false, b1, (short)0, c1, false, false);

        // B2: each lane drops its own 8 activated feats at e0..7 (split-K),
        // activation done packed in f16 after the down-convert.
        #pragma unroll
        for (int i = 0; i < 4; ++i)
            b2.w[i] = h2i(lrelu_pk(d1[2 * i], d1[2 * i + 1]));

        // GEMM2: h2^T[feat][obj]
        v8f d2 = __builtin_amdgcn_wmma_f32_16x16x32_f16(
            false, a2, false, b2.v, (short)0, c2, false, false);

        // Object-sum: packed-f16 xor-tree across the 16 lane-columns, then
        // deposit into B3 column j (lane j keeps feats 0..7, lane j+16 8..15).
        #pragma unroll
        for (int i = 0; i < 4; ++i) {
            h2 p = lrelu_pk(d2[2 * i], d2[2 * i + 1]);
            p = p + ih2(__shfl_xor(h2i(p), 1, 16));
            p = p + ih2(__shfl_xor(h2i(p), 2, 16));
            p = p + ih2(__shfl_xor(h2i(p), 4, 16));
            p = p + ih2(__shfl_xor(h2i(p), 8, 16));
            b3.w[i] = (m == j) ? h2i(p) : b3.w[i];
        }
    }

    // GEMM3: out^T[feat][batch_j] for all 16 batch elements of the tile.
    v8f d3 = __builtin_amdgcn_wmma_f32_16x16x32_f16(
        false, a3, false, b3.v, (short)0, c3, false, false);

    float* op = out + (base + m) * 16 + off;
    #pragma unroll
    for (int v = 0; v < 8; ++v) op[v] = lrelu(d3[v]);
}

extern "C" void kernel_launch(void* const* d_in, const int* in_sizes, int n_in,
                              void* d_out, int out_size, void* d_ws, size_t ws_size,
                              hipStream_t stream) {
    const float* x           = (const float*)d_in[0];
    const float* own_emb     = (const float*)d_in[1];
    const float* agent_emb   = (const float*)d_in[2];
    const float* subgoal_emb = (const float*)d_in[3];
    const float* goal_emb    = (const float*)d_in[4];
    const float* rel_w0      = (const float*)d_in[5];
    const float* rel_b0      = (const float*)d_in[6];
    const float* rel_w1      = (const float*)d_in[7];
    const float* rel_b1      = (const float*)d_in[8];
    const float* mlp_w0      = (const float*)d_in[9];
    const float* mlp_b0      = (const float*)d_in[10];
    float* out = (float*)d_out;

    const int B = in_sizes[0] / 48;          // 262144
    const int ntiles = B / 16;               // 16 batch elements per wave
    const int wavesPerBlock = 8;             // 256 threads
    const int grid = (ntiles + wavesPerBlock - 1) / wavesPerBlock;

    relation_wmma_kernel<<<grid, 256, 0, stream>>>(
        x, own_emb, agent_emb, subgoal_emb, goal_emb,
        rel_w0, rel_b0, rel_w1, rel_b1, mlp_w0, mlp_b0, out, ntiles);
}
